// AssetGraphSAGE_90151363543730
// MI455X (gfx1250) — matrix operand order repaired and checked by reading, not compiled, not run on hardware
//
#include <hip/hip_runtime.h>
#include <hip/hip_bf16.h>

typedef __attribute__((ext_vector_type(16))) __bf16 v16bf;
typedef __attribute__((ext_vector_type(8)))  __bf16 v8bf;
typedef __attribute__((ext_vector_type(8)))  float  v8f;
typedef int v4i __attribute__((vector_size(16)));
typedef __attribute__((address_space(1))) v4i* gas_v4i_p;
typedef __attribute__((address_space(3))) v4i* las_v4i_p;

#define NN 50000
#define NE 800000
#define INC 128
#define HID 256
#define OUTC 128
#define NG 64

#if defined(__HIP_DEVICE_COMPILE__) && __has_builtin(__builtin_amdgcn_global_load_async_to_lds_b128)
#define ASYNC_LDS 1
#else
#define ASYNC_LDS 0
#endif

static __device__ __forceinline__ float half16_sum(float v) {
    v += __shfl_xor(v, 1, 32);
    v += __shfl_xor(v, 2, 32);
    v += __shfl_xor(v, 4, 32);
    v += __shfl_xor(v, 8, 32);
    return v;
}

// ---------------- weight packing: W[K,N] f32 -> WMMA-B fragment order bf16 ----
// packed[((kt*NT+nt)*32 + lane)*16 + j] = W[(kt*32 + (lane>>4)*16 + j)*N + nt*16 + (lane&15)]
__global__ void k_pack_w(const float* __restrict__ W, __bf16* __restrict__ out, int K, int N) {
    int idx = blockIdx.x * blockDim.x + threadIdx.x;
    if (idx >= K * N) return;
    int j    = idx & 15;
    int lane = (idx >> 4) & 31;
    int t    = idx >> 9;            // kt*NT + nt
    int NT   = N >> 4;
    int nt   = t % NT;
    int kt   = t / NT;
    int krow = kt * 32 + (lane >> 4) * 16 + j;
    int ncol = nt * 16 + (lane & 15);
    out[idx] = (__bf16)W[(size_t)krow * N + ncol];
}

__global__ void k_cvt_bf16(const float* __restrict__ in, __bf16* __restrict__ out, int n) {
    int i = blockIdx.x * blockDim.x + threadIdx.x;
    if (i < n) out[i] = (__bf16)in[i];
}
__global__ void k_zero_f32(float* p, int n) {
    int i = blockIdx.x * blockDim.x + threadIdx.x;
    if (i < n) p[i] = 0.f;
}
__global__ void k_zero_u32(unsigned* p, int n) {
    int i = blockIdx.x * blockDim.x + threadIdx.x;
    if (i < n) p[i] = 0u;
}

// ---------------- LDS staging of packed weights ------------------------------
static __device__ __forceinline__ void stage_copy(const __bf16* __restrict__ src,
                                                  __bf16* __restrict__ sdst,
                                                  int nelem, int tid) {
#if ASYNC_LDS
    for (int c = tid * 8; c < nelem; c += 256 * 8) {
        __builtin_amdgcn_global_load_async_to_lds_b128(
            (gas_v4i_p)(uintptr_t)(src + c),
            (las_v4i_p)(unsigned)(uintptr_t)(sdst + c),
            0, 0);
    }
#else
    for (int c = tid * 8; c < nelem; c += 256 * 8) {
        *(v8bf*)(sdst + c) = *(const v8bf*)(src + c);
    }
#endif
}

static __device__ __forceinline__ void wait_async() {
#if ASYNC_LDS
#if __has_builtin(__builtin_amdgcn_s_wait_asynccnt)
    __builtin_amdgcn_s_wait_asynccnt(0);
#else
    asm volatile("s_wait_asynccnt 0x0" ::: "memory");
#endif
#endif
}

// ---------------- GEMM core: per-wave 16-row strip, NT column tiles ----------
// B fragments come from LDS (staged per segment of nkt K-tiles).
template <int NT>
static __device__ __forceinline__ void accum_stage(
    const __bf16* __restrict__ A, int lda, int nkt,
    const __bf16* __restrict__ sW,   // shared-memory base (addrspace inferred after inline)
    int rowA, int khalf, int lane, v8f* acc) {
    const __bf16* pa0 = A + (size_t)rowA * lda + khalf * 16;
    for (int k = 0; k < nkt; ++k) {
        const __bf16* pa = pa0 + k * 32;
        v8bf lo = *(const v8bf*)pa;
        v8bf hi = *(const v8bf*)(pa + 8);
        union { v16bf v; v8bf h[2]; } ua;
        ua.h[0] = lo; ua.h[1] = hi;
        const __bf16* pw = sW + ((size_t)(k * NT) * 32 + lane) * 16;
#pragma unroll
        for (int nt = 0; nt < NT; ++nt) {
            v16bf b = *(const v16bf*)(pw + nt * 512);
            acc[nt] = __builtin_amdgcn_wmma_f32_16x16x32_bf16(
                false, ua.v, false, b, (short)0, acc[nt], false, false);
        }
    }
}

// EPI: 0 = bias+LN+ReLU->bf16 ; 1 = bias+L2norm+LN+ReLU->bf16 (DUAL) ;
//      2 = bias+ReLU->bf16    ; 3 = bias+ReLU->f32 ; 4 = bias->f32
template <int NT, int EPI, bool DUAL>
__global__ __launch_bounds__(256)
void k_gemm(const __bf16* __restrict__ A0, const __bf16* __restrict__ A1,
            const __bf16* __restrict__ A2, int segKT, int totKT, int lda,
            const __bf16* __restrict__ Wp,
            const __bf16* __restrict__ Ab, const __bf16* __restrict__ Wpb,
            int totKTb, int ldab,
            const float* __restrict__ bias,
            const float* __restrict__ ln_g, const float* __restrict__ ln_b,
            void* __restrict__ outv, int nrows)
{
    extern __shared__ __bf16 smem[];
    const int tid    = threadIdx.x;
    const int lane   = tid & 31;
    const int wave   = tid >> 5;
    const int strip  = blockIdx.x * 8 + wave;
    const int strips = nrows >> 4;
    const bool active = strip < strips;      // wave-uniform
    const int row0   = strip << 4;
    const int lane15 = lane & 15;
    const int khalf  = lane >> 4;
    const int rowA   = row0 + lane15;

    v8f acc[NT];
    v8f zero = {0.f, 0.f, 0.f, 0.f, 0.f, 0.f, 0.f, 0.f};
#pragma unroll
    for (int nt = 0; nt < NT; ++nt) acc[nt] = zero;

    const int nStages = totKT / segKT;
    for (int s = 0; s < nStages; ++s) {
        stage_copy(Wp + (size_t)s * segKT * NT * 512, smem, segKT * NT * 512, tid);
        wait_async();
        __syncthreads();
        if (active) {
            const __bf16* Aseg = (s == 0) ? A0 : (s == 1 ? A1 : A2);
            accum_stage<NT>(Aseg, lda, segKT, smem, rowA, khalf, lane, acc);
        }
        __syncthreads();
    }
    if constexpr (DUAL) {
        stage_copy(Wpb, smem, totKTb * NT * 512, tid);
        wait_async();
        __syncthreads();
        if (active)
            accum_stage<NT>(Ab, ldab, totKTb, smem, rowA, khalf, lane, acc);
    }
    if (!active) return;

#pragma unroll
    for (int nt = 0; nt < NT; ++nt) {
        float bv = bias[nt * 16 + lane15];
#pragma unroll
        for (int r = 0; r < 8; ++r) acc[nt][r] += bv;
    }

    if constexpr (EPI == 1) {               // row L2 normalize
#pragma unroll
        for (int r = 0; r < 8; ++r) {
            float s = 0.f;
#pragma unroll
            for (int nt = 0; nt < NT; ++nt) s += acc[nt][r] * acc[nt][r];
            s = half16_sum(s);
            float sc = 1.f / fmaxf(sqrtf(s), 1e-12f);
#pragma unroll
            for (int nt = 0; nt < NT; ++nt) acc[nt][r] *= sc;
        }
    }
    if constexpr (EPI == 0 || EPI == 1) {   // LayerNorm
        const float invN = 1.f / (16.f * NT);
#pragma unroll
        for (int r = 0; r < 8; ++r) {
            float s = 0.f, q = 0.f;
#pragma unroll
            for (int nt = 0; nt < NT; ++nt) { float x = acc[nt][r]; s += x; q += x * x; }
            s = half16_sum(s); q = half16_sum(q);
            float mu  = s * invN;
            float var = q * invN - mu * mu;
            float rs  = rsqrtf(var + 1e-5f);
#pragma unroll
            for (int nt = 0; nt < NT; ++nt) {
                float g = ln_g[nt * 16 + lane15], b2 = ln_b[nt * 16 + lane15];
                acc[nt][r] = (acc[nt][r] - mu) * rs * g + b2;
            }
        }
    }
    if constexpr (EPI <= 3) {               // ReLU
#pragma unroll
        for (int nt = 0; nt < NT; ++nt)
#pragma unroll
            for (int r = 0; r < 8; ++r) acc[nt][r] = fmaxf(acc[nt][r], 0.f);
    }

    const int outN  = NT * 16;
    const int rbase = khalf * 8;
    if constexpr (EPI >= 3) {
        float* out = (float*)outv;
#pragma unroll
        for (int nt = 0; nt < NT; ++nt)
#pragma unroll
            for (int r = 0; r < 8; ++r)
                out[(size_t)(row0 + rbase + r) * outN + nt * 16 + lane15] = acc[nt][r];
    } else {
        __bf16* out = (__bf16*)outv;
#pragma unroll
        for (int nt = 0; nt < NT; ++nt)
#pragma unroll
            for (int r = 0; r < 8; ++r)
                out[(size_t)(row0 + rbase + r) * outN + nt * 16 + lane15] = (__bf16)acc[nt][r];
    }
}

// ---------------- graph aggregation ------------------------------------------
__global__ void k_degree(const int* __restrict__ dst, float* __restrict__ deg, int nE) {
    int i = blockIdx.x * blockDim.x + threadIdx.x;
    if (i < nE) atomicAdd(&deg[dst[i]], 1.f);
}
__global__ void k_invdeg(float* d, int n) {
    int i = blockIdx.x * blockDim.x + threadIdx.x;
    if (i < n) d[i] = 1.f / fmaxf(d[i], 1.f);
}
__global__ void k_scatter(const int* __restrict__ src, const int* __restrict__ dst,
                          const __bf16* __restrict__ h, float* __restrict__ nbr, int nE) {
    long long idx = (long long)blockIdx.x * blockDim.x + threadIdx.x;
    int e = (int)(idx >> 6);
    if (e >= nE) return;
    int j = ((int)idx & 63) << 2;
    int s = src[e], d = dst[e];
    const __bf16* hp = h + (size_t)s * HID + j;
    float* np = nbr + (size_t)d * HID + j;
    atomicAdd(np + 0, (float)hp[0]);
    atomicAdd(np + 1, (float)hp[1]);
    atomicAdd(np + 2, (float)hp[2]);
    atomicAdd(np + 3, (float)hp[3]);
}
__global__ void k_scale_cvt(const float* __restrict__ nbr, const float* __restrict__ inv,
                            __bf16* __restrict__ out, int n4) {
    int idx = blockIdx.x * blockDim.x + threadIdx.x;
    if (idx >= n4) return;
    int base = idx << 2;
    float s = inv[base >> 8];
#pragma unroll
    for (int k = 0; k < 4; ++k) out[base + k] = (__bf16)(nbr[base + k] * s);
}

// ---------------- attention pooling ------------------------------------------
__global__ void k_gate2(const __bf16* __restrict__ gtmp, const float* __restrict__ g2W,
                        const float* __restrict__ g2b, float* __restrict__ gate, int n) {
    int i = blockIdx.x * blockDim.x + threadIdx.x;
    if (i >= n) return;
    float a = g2b[0];
    const __bf16* p = gtmp + (size_t)i * (HID / 2);
#pragma unroll 8
    for (int j = 0; j < HID / 2; ++j) a += (float)p[j] * g2W[j];
    gate[i] = a;
}
static __device__ __forceinline__ unsigned fflip(float f) {
    unsigned u = __float_as_uint(f);
    return (u & 0x80000000u) ? ~u : (u | 0x80000000u);
}
static __device__ __forceinline__ float funflip(unsigned u) {
    return (u & 0x80000000u) ? __uint_as_float(u ^ 0x80000000u) : __uint_as_float(~u);
}
__global__ void k_segmax(const float* __restrict__ gate, const int* __restrict__ batch,
                         unsigned* __restrict__ gmax, int n) {
    int i = blockIdx.x * blockDim.x + threadIdx.x;
    if (i < n) atomicMax(&gmax[batch[i]], fflip(gate[i]));
}
__global__ void k_expsum(float* __restrict__ gate, const int* __restrict__ batch,
                         const unsigned* __restrict__ gmax, float* __restrict__ denom, int n) {
    int i = blockIdx.x * blockDim.x + threadIdx.x;
    if (i >= n) return;
    int b = batch[i];
    float e = expf(gate[i] - funflip(gmax[b]));
    gate[i] = e;
    atomicAdd(&denom[b], e);
}
__global__ void k_pool(const float* __restrict__ e, const float* __restrict__ denom,
                       const int* __restrict__ batch, const float* __restrict__ ht,
                       float* __restrict__ pooled, int n4) {
    int idx = blockIdx.x * blockDim.x + threadIdx.x;
    if (idx >= n4) return;
    int base = idx << 2;
    int node = base >> 8;
    int b = batch[node];
    float a = e[node] / denom[b];
#pragma unroll
    for (int k = 0; k < 4; ++k)
        atomicAdd(&pooled[(size_t)b * HID + (base & 255) + k], a * ht[base + k]);
}

// ---------------- driver ------------------------------------------------------
extern "C" void kernel_launch(void* const* d_in, const int* in_sizes, int n_in,
                              void* d_out, int out_size, void* d_ws, size_t ws_size,
                              hipStream_t stream) {
    (void)in_sizes; (void)n_in; (void)out_size; (void)ws_size;
    const float* x     = (const float*)d_in[0];
    const int*   src   = (const int*)d_in[1];
    const int*   dst   = src + NE;
    const int*   batch = (const int*)d_in[2];
    const float* in_W  = (const float*)d_in[3];
    const float* in_b  = (const float*)d_in[4];
    const float* in_g  = (const float*)d_in[5];
    const float* in_bb = (const float*)d_in[6];
    const float *Wl[3], *bl[3], *Wr[3], *lg[3], *lb[3];
    for (int i = 0; i < 3; ++i) {
        Wl[i] = (const float*)d_in[7 + 5 * i];
        bl[i] = (const float*)d_in[8 + 5 * i];
        Wr[i] = (const float*)d_in[9 + 5 * i];
        lg[i] = (const float*)d_in[10 + 5 * i];
        lb[i] = (const float*)d_in[11 + 5 * i];
    }
    const float* jkW  = (const float*)d_in[22];
    const float* jkb  = (const float*)d_in[23];
    const float* jkg  = (const float*)d_in[24];
    const float* jkbb = (const float*)d_in[25];
    const float* g1W  = (const float*)d_in[26];
    const float* g1b  = (const float*)d_in[27];
    const float* g2W  = (const float*)d_in[28];
    const float* g2b  = (const float*)d_in[29];
    const float* pnW  = (const float*)d_in[30];
    const float* pnb  = (const float*)d_in[31];
    const float* p1W  = (const float*)d_in[32];
    const float* p1b  = (const float*)d_in[33];
    const float* prg  = (const float*)d_in[34];
    const float* prb  = (const float*)d_in[35];
    const float* p2W  = (const float*)d_in[36];
    const float* p2b  = (const float*)d_in[37];

    // workspace carve (256B aligned)
    char* ws = (char*)d_ws;
    size_t o = 0;
    auto carve = [&](size_t bytes) { char* r = ws + o; o = (o + bytes + 255) & ~(size_t)255; return r; };
    __bf16* xB     = (__bf16*)carve((size_t)NN * INC * 2);
    __bf16* h0     = (__bf16*)carve((size_t)NN * HID * 2);   // in-proj out; reused as hJK
    __bf16* L0     = (__bf16*)carve((size_t)NN * HID * 2);
    __bf16* L1     = (__bf16*)carve((size_t)NN * HID * 2);
    __bf16* L2     = (__bf16*)carve((size_t)NN * HID * 2);
    float*  nbrF   = (float*) carve((size_t)NN * HID * 4);   // reused as ht
    __bf16* nbrB   = (__bf16*)carve((size_t)NN * HID * 2);
    __bf16* gtmp   = (__bf16*)carve((size_t)NN * (HID / 2) * 2);
    float*  gate   = (float*) carve((size_t)NN * 4);
    float*  invd   = (float*) carve((size_t)NN * 4);
    unsigned* gmax = (unsigned*)carve(NG * 4);
    float*  denom  = (float*) carve(NG * 4);
    float*  pooled = (float*) carve((size_t)NG * HID * 4);
    __bf16* pooledB= (__bf16*)carve((size_t)NG * HID * 2);
    __bf16* zB     = (__bf16*)carve((size_t)NG * HID * 2);
    __bf16* pInW   = (__bf16*)carve((size_t)INC * HID * 2);
    __bf16* pWl[3], *pWr[3];
    for (int i = 0; i < 3; ++i) {
        pWl[i] = (__bf16*)carve((size_t)HID * HID * 2);
        pWr[i] = (__bf16*)carve((size_t)HID * HID * 2);
    }
    __bf16* pJk = (__bf16*)carve((size_t)3 * HID * HID * 2);
    __bf16* pG1 = (__bf16*)carve((size_t)HID * (HID / 2) * 2);
    __bf16* pPn = (__bf16*)carve((size_t)HID * HID * 2);
    __bf16* pP1 = (__bf16*)carve((size_t)HID * HID * 2);
    __bf16* pP2 = (__bf16*)carve((size_t)HID * OUTC * 2);

    auto cdiv = [](long long a, long long b) { return (unsigned)((a + b - 1) / b); };
    auto pack = [&](const float* W, __bf16* out, int K, int N) {
        k_pack_w<<<cdiv((long long)K * N, 256), 256, 0, stream>>>(W, out, K, N);
    };

    // 0) pack weights + convert x
    pack(in_W, pInW, INC, HID);
    for (int i = 0; i < 3; ++i) { pack(Wl[i], pWl[i], HID, HID); pack(Wr[i], pWr[i], HID, HID); }
    pack(jkW, pJk, 3 * HID, HID);
    pack(g1W, pG1, HID, HID / 2);
    pack(pnW, pPn, HID, HID);
    pack(p1W, pP1, HID, HID);
    pack(p2W, pP2, HID, OUTC);
    k_cvt_bf16<<<cdiv((long long)NN * INC, 256), 256, 0, stream>>>(x, xB, NN * INC);

    // 1) degrees
    k_zero_f32<<<cdiv(NN, 256), 256, 0, stream>>>(invd, NN);
    k_degree<<<cdiv(NE, 256), 256, 0, stream>>>(dst, invd, NE);
    k_invdeg<<<cdiv(NN, 256), 256, 0, stream>>>(invd, NN);

    const int strips  = NN / 16;
    const int gblocks = (strips + 7) / 8;
    const unsigned lds64  = 64 * 1024;    // 4 ktiles x 16 ntiles or 8 ktiles x 8 ntiles
    const unsigned lds128 = 128 * 1024;   // 8 ktiles x 16 ntiles

    // 2) input projection: h0 = relu(LN(x @ in_W + b))
    k_gemm<16, 0, false><<<gblocks, 256, lds64, stream>>>(
        xB, xB, xB, INC / 32, INC / 32, INC, pInW,
        nullptr, nullptr, 0, 0, in_b, in_g, in_bb, h0, NN);

    // 3) SAGE layers
    __bf16* hprev = h0;
    __bf16* Ls[3] = {L0, L1, L2};
    for (int i = 0; i < 3; ++i) {
        k_zero_f32<<<cdiv((long long)NN * HID, 256), 256, 0, stream>>>(nbrF, NN * HID);
        k_scatter<<<cdiv((long long)NE * 64, 256), 256, 0, stream>>>(src, dst, hprev, nbrF, NE);
        k_scale_cvt<<<cdiv((long long)NN * HID / 4, 256), 256, 0, stream>>>(
            nbrF, invd, nbrB, NN * HID / 4);
        k_gemm<16, 1, true><<<gblocks, 256, lds128, stream>>>(
            nbrB, nbrB, nbrB, HID / 32, HID / 32, HID, pWl[i],
            hprev, pWr[i], HID / 32, HID, bl[i], lg[i], lb[i], Ls[i], NN);
        hprev = Ls[i];
    }

    // 4) jumping knowledge: hJK = relu(LN(concat(L0,L1,L2) @ jk_W + b))  (reuse h0)
    k_gemm<16, 0, false><<<gblocks, 256, lds128, stream>>>(
        L0, L1, L2, HID / 32, 3 * HID / 32, HID, pJk,
        nullptr, nullptr, 0, 0, jkb, jkg, jkbb, h0, NN);

    // 5) gate = relu(hJK @ g1 + b) @ g2 + b
    k_gemm<8, 2, false><<<gblocks, 256, lds64, stream>>>(
        h0, h0, h0, HID / 32, HID / 32, HID, pG1,
        nullptr, nullptr, 0, 0, g1b, nullptr, nullptr, gtmp, NN);
    k_gate2<<<cdiv(NN, 256), 256, 0, stream>>>(gtmp, g2W, g2b, gate, NN);

    // 6) ht = relu(hJK @ pn_W + b)  (f32, reuse nbrF)
    k_gemm<16, 3, false><<<gblocks, 256, lds128, stream>>>(
        h0, h0, h0, HID / 32, HID / 32, HID, pPn,
        nullptr, nullptr, 0, 0, pnb, nullptr, nullptr, nbrF, NN);

    // 7) segment softmax + weighted pooling
    k_zero_u32<<<1, NG, 0, stream>>>(gmax, NG);
    k_segmax<<<cdiv(NN, 256), 256, 0, stream>>>(gate, batch, gmax, NN);
    k_zero_f32<<<1, NG, 0, stream>>>(denom, NG);
    k_expsum<<<cdiv(NN, 256), 256, 0, stream>>>(gate, batch, gmax, denom, NN);
    k_zero_f32<<<cdiv(NG * HID, 256), 256, 0, stream>>>(pooled, NG * HID);
    k_pool<<<cdiv((long long)NN * HID / 4, 256), 256, 0, stream>>>(
        gate, denom, batch, nbrF, pooled, NN * HID / 4);
    k_cvt_bf16<<<cdiv(NG * HID, 256), 256, 0, stream>>>(pooled, pooledB, NG * HID);

    // 8) projection head (M = 64)
    k_gemm<16, 0, false><<<1, 256, lds128, stream>>>(
        pooledB, pooledB, pooledB, HID / 32, HID / 32, HID, pP1,
        nullptr, nullptr, 0, 0, p1b, prg, prb, zB, NG);
    k_gemm<8, 4, false><<<1, 256, lds64, stream>>>(
        zB, zB, zB, HID / 32, HID / 32, HID, pP2,
        nullptr, nullptr, 0, 0, p2b, nullptr, nullptr, d_out, NG);
}
